// GeneratorSeq2Seq_31207232373358
// MI455X (gfx1250) — matrix operand order repaired and checked by reading, not compiled, not run on hardware
//
#include <hip/hip_runtime.h>

// ---------------------------------------------------------------------------
// Seq2seq (2-layer LSTM enc/dec + attention) for MI455X (gfx1250, wave32).
//  * Weights packed per call into f16 WMMA B-operand layout (L2-resident).
//  * Activations produced directly in f16 WMMA A-operand layout by the
//    producer kernels, so the GEMM inner loop is pure loads + v_wmma.
//  * GEMM K-loop software-pipelined in two select-free phases (A0 sweep,
//    pipelined bridge, A1 sweep) so next-iteration loads overlap WMMAs.
// ---------------------------------------------------------------------------

typedef __attribute__((ext_vector_type(16))) _Float16 v16h;
typedef __attribute__((ext_vector_type(8)))  float    v8f;

#define HID     1024
#define BATCH   64
#define SEQT    128
#define VOCAB   512
#define GATES4H 4096
#define BH      (BATCH * HID)
#define KTA     32          // K-tiles (of 32) per [64,1024] activation
#define PACK_THREADS 4096   // 4 mtiles * 32 ktiles * 32 lanes

// A-fragment tile addressing (per [64,1024] activation):
//   tile (mtile, kt), lane L: row b = mtile*16 + (L&15), g = L>>4,
//   halfs 0..7  hold k = kt*32 + g*8 + (0..7)
//   halfs 8..15 hold k = kt*32 + 16 + g*8 + (0..7)
//   stored contiguously at ((mtile*KTA + kt)*32 + L)*16 halfs.
__device__ inline void pk_decode(int tid, int& b, int& kb)
{
    int lane  = tid & 31;
    int kt    = (tid >> 5) & 31;
    int mtile = tid >> 10;
    b  = mtile * 16 + (lane & 15);
    kb = kt * 32 + (lane >> 4) * 8;
}

// ---------------------------------------------------------------------------
// Weight packing: f32 row-major [N,K] (optionally split at K0 between two
// sources) -> f16 WMMA B-operand tiles: lane L: n = ntile*16 + (L&15),
// halfs 0..7 hold k = ktile*32 + (L>>4)*8 + j, halfs 8..15 hold k = ktile*32 + 16 + (L>>4)*8 + (j-8).
// ---------------------------------------------------------------------------
__global__ void pack_weights_kernel(const float* __restrict__ W0,
                                    const float* __restrict__ W1,
                                    int ld0, int ld1, int K0,
                                    int Ntiles, int KT,
                                    _Float16* __restrict__ out)
{
    int tid = blockIdx.x * blockDim.x + threadIdx.x;   // one per (ntile,kt,lane)
    int total = Ntiles * KT * 32;
    if (tid >= total) return;
    int lane  = tid & 31;
    int kt    = (tid >> 5) % KT;
    int ntile = tid / (32 * KT);

    int n   = ntile * 16 + (lane & 15);
    int kin = (lane >> 4) * 8;
    _Float16* dst = out + (size_t)tid * 16;
#pragma unroll
    for (int j = 0; j < 16; ++j) {
        int k = kt * 32 + kin + (j & 7) + ((j >> 3) << 4);
        float v = (k < K0) ? W0[(size_t)n * ld0 + k]
                           : W1[(size_t)n * ld1 + (k - K0)];
        dst[j] = (_Float16)v;
    }
}

// ---------------------------------------------------------------------------
// WMMA GEMM: C[64,N] = concat(A0,A1) * Bpacked (+bias, +relu).
// A0/A1 are pre-packed f16 A-fragments of [64,1024] activations.
// Block = 8 waves; each wave owns one 16-wide N tile and all 4 M tiles.
// mode: 0 = f32 C, 1 = f32 C + relu, 2 = packed f16 A-layout C + relu.
// ---------------------------------------------------------------------------
__device__ inline v8f wmma_acc(v16h a, v16h b, v8f c)
{
    return __builtin_amdgcn_wmma_f32_16x16x32_f16(false, a, false, b,
                                                  (short)0, c, false, false);
}

#define LOAD_FRAG(BV, A0V, A1V, A2V, A3V, BP, AP)                 \
    do {                                                          \
        (BV)  = *(const v16h*)(BP);                               \
        (A0V) = *(const v16h*)(AP);                               \
        (A1V) = *(const v16h*)((AP) + 16384);                     \
        (A2V) = *(const v16h*)((AP) + 32768);                     \
        (A3V) = *(const v16h*)((AP) + 49152);                     \
    } while (0)

#define WMMA4(ACC, CB, CA0, CA1, CA2, CA3)                        \
    do {                                                          \
        (ACC)[0] = wmma_acc((CA0), (CB), (ACC)[0]);               \
        (ACC)[1] = wmma_acc((CA1), (CB), (ACC)[1]);               \
        (ACC)[2] = wmma_acc((CA2), (CB), (ACC)[2]);               \
        (ACC)[3] = wmma_acc((CA3), (CB), (ACC)[3]);               \
    } while (0)

__global__ __launch_bounds__(256)
void wmma_gemm_kernel(const _Float16* __restrict__ pA0,
                      const _Float16* __restrict__ pA1,
                      const _Float16* __restrict__ Bp,
                      const float* __restrict__ bias,
                      void* __restrict__ Cout,
                      int N, int KT0, int KT1, int mode)
{
    const int lane  = threadIdx.x & 31;
    const int wid   = threadIdx.x >> 5;
    const int ntile = blockIdx.x * 8 + wid;
    if (ntile * 16 >= N) return;          // wave-uniform: EXEC stays all-ones

    const int KT = KT0 + KT1;
    v8f acc[4] = {};
    const _Float16* bptr = Bp + (size_t)ntile * KT * 512 + (size_t)lane * 16;
    const _Float16* a0b  = pA0 + (size_t)lane * 16;
    const _Float16* a1b  = pA1 ? pA1 + (size_t)lane * 16 : a0b;

    // ---- software-pipelined, select-free K loop ----
    v16h cb, ca0, ca1, ca2, ca3;
    LOAD_FRAG(cb, ca0, ca1, ca2, ca3, bptr, a0b);
    bptr += 512;

    // phase 1: "next" fragments come from A0 (kt = 1 .. KT0-1)
    const _Float16* ap = a0b + 512;
    for (int kt = 1; kt < KT0; ++kt, ap += 512) {
        v16h nb, na0, na1, na2, na3;
        LOAD_FRAG(nb, na0, na1, na2, na3, bptr, ap);
        __builtin_prefetch(bptr + 8 * 512, 0, 1);   // weight stream hint
        bptr += 512;
        WMMA4(acc, cb, ca0, ca1, ca2, ca3);
        cb = nb; ca0 = na0; ca1 = na1; ca2 = na2; ca3 = na3;
    }
    // pipelined bridge + phase 2: "next" fragments come from A1
    ap = a1b;
    for (int kt = 0; kt < KT1; ++kt, ap += 512) {
        v16h nb, na0, na1, na2, na3;
        LOAD_FRAG(nb, na0, na1, na2, na3, bptr, ap);
        __builtin_prefetch(bptr + 8 * 512, 0, 1);
        bptr += 512;
        WMMA4(acc, cb, ca0, ca1, ca2, ca3);
        cb = nb; ca0 = na0; ca1 = na1; ca2 = na2; ca3 = na3;
    }
    // epilogue: last fragment
    WMMA4(acc, cb, ca0, ca1, ca2, ca3);

    const int g = lane >> 4;
    const int r = lane & 15;
    const int n = ntile * 16 + r;
    const float bv = bias ? bias[n] : 0.0f;

    if (mode <= 1) {
        float* C = (float*)Cout;
#pragma unroll
        for (int mt = 0; mt < 4; ++mt)
#pragma unroll
            for (int i = 0; i < 8; ++i) {
                float v = acc[mt][i] + bv;
                if (mode == 1) v = fmaxf(v, 0.0f);
                C[(size_t)(mt * 16 + g * 8 + i) * N + n] = v;
            }
    } else {
        // scatter into f16 A-fragment layout (row b = m, col k = n); relu.
        _Float16* P = (_Float16*)Cout;
        const int kin  = n & 31;
        const int ga   = (kin >> 3) & 1;
        const int e    = (kin & 7) + ((kin >> 4) << 3);
        const int ktA  = n >> 5;
#pragma unroll
        for (int mt = 0; mt < 4; ++mt)
#pragma unroll
            for (int i = 0; i < 8; ++i) {
                int m = mt * 16 + g * 8 + i;
                float v = fmaxf(acc[mt][i] + bv, 0.0f);
                int laneA = (m & 15) + 16 * ga;
                size_t offp = (((size_t)(m >> 4) * KTA + ktA) * 32 + laneA) * 16 + e;
                P[offp] = (_Float16)v;
            }
    }
}

// ---------------------------------------------------------------------------
// Producers: emit activations directly as packed f16 A-fragments.
// ---------------------------------------------------------------------------
__device__ inline float sigf(float x) { return 1.0f / (1.0f + __expf(-x)); }

__device__ inline v16h pack_from_rows(const float* __restrict__ src, int kb)
{
    float4 q0 = *(const float4*)(src + kb);
    float4 q1 = *(const float4*)(src + kb + 4);
    float4 q2 = *(const float4*)(src + kb + 16);
    float4 q3 = *(const float4*)(src + kb + 20);
    v16h a;
    a[0]=(_Float16)q0.x;  a[1]=(_Float16)q0.y;  a[2]=(_Float16)q0.z;  a[3]=(_Float16)q0.w;
    a[4]=(_Float16)q1.x;  a[5]=(_Float16)q1.y;  a[6]=(_Float16)q1.z;  a[7]=(_Float16)q1.w;
    a[8]=(_Float16)q2.x;  a[9]=(_Float16)q2.y;  a[10]=(_Float16)q2.z; a[11]=(_Float16)q2.w;
    a[12]=(_Float16)q3.x; a[13]=(_Float16)q3.y; a[14]=(_Float16)q3.z; a[15]=(_Float16)q3.w;
    return a;
}

__global__ void embed_pack_kernel(const int* __restrict__ tokens,   // [B] row
                                  const float* __restrict__ emb,
                                  _Float16* __restrict__ xp)
{
    int tid = blockIdx.x * blockDim.x + threadIdx.x;
    if (tid >= PACK_THREADS) return;
    int b, kb; pk_decode(tid, b, kb);
    const float* src = emb + (size_t)tokens[b] * HID;
    *(v16h*)(xp + (size_t)tid * 16) = pack_from_rows(src, kb);
}

__global__ void dec_embed_pack_kernel(const int* __restrict__ input0,
                                      const int* __restrict__ target,
                                      const int* __restrict__ tf_flag,
                                      const int* __restrict__ prev_tok,
                                      int di,
                                      const float* __restrict__ emb,
                                      _Float16* __restrict__ xp)
{
    int tid = blockIdx.x * blockDim.x + threadIdx.x;
    if (tid >= PACK_THREADS) return;
    int b, kb; pk_decode(tid, b, kb);
    int tok;
    if (di == 0)           tok = input0[b];
    else if (*tf_flag)     tok = target[(size_t)(di - 1) * BATCH + b];
    else                   tok = prev_tok[b];
    const float* src = emb + (size_t)tok * HID;
    *(v16h*)(xp + (size_t)tid * 16) = pack_from_rows(src, kb);
}

// LSTM pointwise: consumes f32 gates [B,4H], updates c (f32), writes packed
// f16 h, optional f32 copy (encoder outputs).
__global__ void lstm_pointwise_pack_kernel(const float* __restrict__ gates,
                                           const float* __restrict__ bih,
                                           const float* __restrict__ bhh,
                                           float* __restrict__ c,
                                           _Float16* __restrict__ hpack,
                                           float* __restrict__ hcopy)
{
    int tid = blockIdx.x * blockDim.x + threadIdx.x;
    if (tid >= PACK_THREADS) return;
    int b, kb; pk_decode(tid, b, kb);
    const float* gr = gates + (size_t)b * GATES4H;
    float* crow = c + (size_t)b * HID;
    float* hrow = hcopy ? hcopy + (size_t)b * HID : nullptr;
    v16h hv;
#pragma unroll
    for (int half = 0; half < 2; ++half) {
#pragma unroll
        for (int jj = 0; jj < 8; ++jj) {
            int k = kb + half * 16 + jj;
            float gi = gr[k]           + bih[k]           + bhh[k];
            float gf = gr[HID + k]     + bih[HID + k]     + bhh[HID + k];
            float gg = gr[2*HID + k]   + bih[2*HID + k]   + bhh[2*HID + k];
            float go = gr[3*HID + k]   + bih[3*HID + k]   + bhh[3*HID + k];
            float cn = sigf(gf) * crow[k] + sigf(gi) * tanhf(gg);
            crow[k] = cn;
            float hn = sigf(go) * tanhf(cn);
            hv[half * 8 + jj] = (_Float16)hn;
            if (hrow) hrow[k] = hn;
        }
    }
    *(v16h*)(hpack + (size_t)tid * 16) = hv;
}

// attn_applied[b,k] = sum_t w[b,t] * enc_outs[t,b,k] -> packed f16 fragments
__global__ void attn_apply_pack_kernel(const float* __restrict__ w,
                                       const float* __restrict__ enc_outs,
                                       _Float16* __restrict__ outp)
{
    int tid = blockIdx.x * blockDim.x + threadIdx.x;
    if (tid >= PACK_THREADS) return;
    int b, kb; pk_decode(tid, b, kb);
    const float* wr = w + (size_t)b * SEQT;
    float acc[16];
#pragma unroll
    for (int i = 0; i < 16; ++i) acc[i] = 0.0f;
    for (int t = 0; t < SEQT; ++t) {
        float wt = wr[t];
        const float* er = enc_outs + ((size_t)t * BATCH + b) * HID + kb;
        float4 q0 = *(const float4*)(er);
        float4 q1 = *(const float4*)(er + 4);
        float4 q2 = *(const float4*)(er + 16);
        float4 q3 = *(const float4*)(er + 20);
        acc[0]  = fmaf(wt, q0.x, acc[0]);  acc[1]  = fmaf(wt, q0.y, acc[1]);
        acc[2]  = fmaf(wt, q0.z, acc[2]);  acc[3]  = fmaf(wt, q0.w, acc[3]);
        acc[4]  = fmaf(wt, q1.x, acc[4]);  acc[5]  = fmaf(wt, q1.y, acc[5]);
        acc[6]  = fmaf(wt, q1.z, acc[6]);  acc[7]  = fmaf(wt, q1.w, acc[7]);
        acc[8]  = fmaf(wt, q2.x, acc[8]);  acc[9]  = fmaf(wt, q2.y, acc[9]);
        acc[10] = fmaf(wt, q2.z, acc[10]); acc[11] = fmaf(wt, q2.w, acc[11]);
        acc[12] = fmaf(wt, q3.x, acc[12]); acc[13] = fmaf(wt, q3.y, acc[13]);
        acc[14] = fmaf(wt, q3.z, acc[14]); acc[15] = fmaf(wt, q3.w, acc[15]);
    }
    v16h a;
#pragma unroll
    for (int i = 0; i < 16; ++i) a[i] = (_Float16)acc[i];
    *(v16h*)(outp + (size_t)tid * 16) = a;
}

// ---------------------------------------------------------------------------
// Row-wise kernels (one wave per row)
// ---------------------------------------------------------------------------
__global__ void softmax128_kernel(const float* __restrict__ in, float* __restrict__ out)
{
    int lane = threadIdx.x & 31, wid = threadIdx.x >> 5;
    int row = blockIdx.x * 4 + wid;
    if (row >= BATCH) return;
    float v[4], mx = -3.0e38f;
#pragma unroll
    for (int i = 0; i < 4; ++i) { v[i] = in[(size_t)row * SEQT + lane + 32 * i]; mx = fmaxf(mx, v[i]); }
#pragma unroll
    for (int off = 16; off >= 1; off >>= 1) mx = fmaxf(mx, __shfl_xor(mx, off));
    float s = 0.0f;
#pragma unroll
    for (int i = 0; i < 4; ++i) { v[i] = __expf(v[i] - mx); s += v[i]; }
#pragma unroll
    for (int off = 16; off >= 1; off >>= 1) s += __shfl_xor(s, off);
    float inv = 1.0f / s;
#pragma unroll
    for (int i = 0; i < 4; ++i) out[(size_t)row * SEQT + lane + 32 * i] = v[i] * inv;
}

__global__ void logsoftmax_kernel(const float* __restrict__ logits,
                                  float* __restrict__ out, int di)
{
    int lane = threadIdx.x & 31, wid = threadIdx.x >> 5;
    int row = blockIdx.x * 4 + wid;
    if (row >= BATCH) return;
    float v[16], mx = -3.0e38f;
#pragma unroll
    for (int i = 0; i < 16; ++i) { v[i] = logits[(size_t)row * VOCAB + lane + 32 * i]; mx = fmaxf(mx, v[i]); }
#pragma unroll
    for (int off = 16; off >= 1; off >>= 1) mx = fmaxf(mx, __shfl_xor(mx, off));
    float s = 0.0f;
#pragma unroll
    for (int i = 0; i < 16; ++i) s += __expf(v[i] - mx);
#pragma unroll
    for (int off = 16; off >= 1; off >>= 1) s += __shfl_xor(s, off);
    float lse = mx + __logf(s);
    float* orow = out + ((size_t)row * SEQT + di) * VOCAB;
#pragma unroll
    for (int i = 0; i < 16; ++i) orow[lane + 32 * i] = v[i] - lse;
}

__global__ void argmax_kernel(const float* __restrict__ logits, int* __restrict__ tok)
{
    int lane = threadIdx.x & 31, wid = threadIdx.x >> 5;
    int row = blockIdx.x * 4 + wid;
    if (row >= BATCH) return;
    float bv = -3.0e38f; int bi = 0;
#pragma unroll
    for (int i = 0; i < 16; ++i) {
        int c = lane + 32 * i;
        float x = logits[(size_t)row * VOCAB + c];
        if (x > bv) { bv = x; bi = c; }
    }
#pragma unroll
    for (int off = 16; off >= 1; off >>= 1) {
        float ov = __shfl_xor(bv, off);
        int   oi = __shfl_xor(bi, off);
        if (ov > bv || (ov == bv && oi < bi)) { bv = ov; bi = oi; }
    }
    if (lane == 0) tok[row] = bi;
}

__global__ void zero_kernel(float* __restrict__ p, int n)
{
    int idx = blockIdx.x * blockDim.x + threadIdx.x;
    if (idx < n) p[idx] = 0.0f;
}

// ---------------------------------------------------------------------------
// Host-side orchestration
// ---------------------------------------------------------------------------
extern "C" void kernel_launch(void* const* d_in, const int* in_sizes, int n_in,
                              void* d_out, int out_size, void* d_ws, size_t ws_size,
                              hipStream_t stream)
{
    const int*   input_tensor  = (const int*)  d_in[0];
    const int*   target_tensor = (const int*)  d_in[1];
    const int*   tf_flag       = (const int*)  d_in[4];
    const float* enc_emb       = (const float*)d_in[5];
    const float* enc_Wih       = (const float*)d_in[6];
    const float* enc_Whh       = (const float*)d_in[7];
    const float* enc_bih       = (const float*)d_in[8];
    const float* enc_bhh       = (const float*)d_in[9];
    const float* dec_emb       = (const float*)d_in[10];
    const float* Wattn         = (const float*)d_in[11];
    const float* battn         = (const float*)d_in[12];
    const float* Wcomb         = (const float*)d_in[13];
    const float* bcomb         = (const float*)d_in[14];
    const float* dec_Wih       = (const float*)d_in[15];
    const float* dec_Whh       = (const float*)d_in[16];
    const float* dec_bih       = (const float*)d_in[17];
    const float* dec_bhh       = (const float*)d_in[18];
    const float* Wout          = (const float*)d_in[19];
    const float* bout          = (const float*)d_in[20];
    float* out = (float*)d_out;

    uint8_t* ws = (uint8_t*)d_ws;
    size_t off = 0;
    auto alloc = [&](size_t bytes) -> void* {
        void* p = ws + off;
        off = (off + bytes + 255) & ~(size_t)255;
        return p;
    };
    const size_t LSTM_PW = (size_t)GATES4H * 2048 * sizeof(_Float16);
    _Float16* pwEnc0 = (_Float16*)alloc(LSTM_PW);
    _Float16* pwEnc1 = (_Float16*)alloc(LSTM_PW);
    _Float16* pwDec0 = (_Float16*)alloc(LSTM_PW);
    _Float16* pwDec1 = (_Float16*)alloc(LSTM_PW);
    _Float16* pwAttn = (_Float16*)alloc((size_t)SEQT  * 2048 * sizeof(_Float16));
    _Float16* pwComb = (_Float16*)alloc((size_t)HID   * 2048 * sizeof(_Float16));
    _Float16* pwOut  = (_Float16*)alloc((size_t)VOCAB * 1024 * sizeof(_Float16));
    float*    enc_outs = (float*)alloc((size_t)SEQT * BH * sizeof(float));   // [T,B,H]
    float*    c0     = (float*)alloc((size_t)BH * sizeof(float));
    float*    c1     = (float*)alloc((size_t)BH * sizeof(float));
    _Float16* pH0    = (_Float16*)alloc((size_t)BH * sizeof(_Float16));
    _Float16* pH1    = (_Float16*)alloc((size_t)BH * sizeof(_Float16));
    _Float16* pEmb   = (_Float16*)alloc((size_t)BH * sizeof(_Float16));
    _Float16* pAttn  = (_Float16*)alloc((size_t)BH * sizeof(_Float16));   // attn_applied
    _Float16* pXcomb = (_Float16*)alloc((size_t)BH * sizeof(_Float16));
    float*    gates   = (float*)alloc((size_t)BATCH * GATES4H * sizeof(float));
    float*    attnlog = (float*)alloc((size_t)BATCH * SEQT * sizeof(float));
    float*    attnw   = (float*)alloc((size_t)BATCH * SEQT * sizeof(float));
    float*    logits  = (float*)alloc((size_t)BATCH * VOCAB * sizeof(float));
    int*      prevtok = (int*)  alloc(256);

    const int TPB = 256;
    const int PACK_BLOCKS = PACK_THREADS / TPB;   // 16

    // --- pack weights into f16 WMMA-B layout ---
    auto pack = [&](const float* W0, const float* W1, int ld0, int ld1,
                    int K0, int N, int K, _Float16* dst) {
        int KT = K / 32;
        int total = (N / 16) * KT * 32;
        pack_weights_kernel<<<(total + TPB - 1) / TPB, TPB, 0, stream>>>(
            W0, W1, ld0, ld1, K0, N / 16, KT, dst);
    };
    const size_t WSTRIDE = (size_t)GATES4H * HID;
    pack(enc_Wih,           enc_Whh,           HID, HID, HID, GATES4H, 2048, pwEnc0);
    pack(enc_Wih + WSTRIDE, enc_Whh + WSTRIDE, HID, HID, HID, GATES4H, 2048, pwEnc1);
    pack(dec_Wih,           dec_Whh,           HID, HID, HID, GATES4H, 2048, pwDec0);
    pack(dec_Wih + WSTRIDE, dec_Whh + WSTRIDE, HID, HID, HID, GATES4H, 2048, pwDec1);
    pack(Wattn, nullptr, 2048, 0, 2048, SEQT,  2048, pwAttn);
    pack(Wcomb, nullptr, 2048, 0, 2048, HID,   2048, pwComb);
    pack(Wout,  nullptr, HID,  0, HID,  VOCAB, 1024, pwOut);

    // zero initial state: c0,c1 (f32) + pH0,pH1 (f16) are contiguous = 3*BH floats
    zero_kernel<<<(3 * BH + TPB - 1) / TPB, TPB, 0, stream>>>(c0, 3 * BH);

    auto gemm = [&](const _Float16* A0, const _Float16* A1, const _Float16* Bp,
                    const float* bias, void* C, int N, int KT1, int mode) {
        int blocks = (N / 16 + 7) / 8;
        wmma_gemm_kernel<<<blocks, 256, 0, stream>>>(A0, A1, Bp, bias, C,
                                                     N, KTA, KT1, mode);
    };

    // ---------------- Encoder ----------------
    for (int t = 0; t < SEQT; ++t) {
        embed_pack_kernel<<<PACK_BLOCKS, TPB, 0, stream>>>(
            input_tensor + (size_t)t * BATCH, enc_emb, pEmb);
        gemm(pEmb, pH0, pwEnc0, nullptr, gates, GATES4H, KTA, 0);
        lstm_pointwise_pack_kernel<<<PACK_BLOCKS, TPB, 0, stream>>>(
            gates, enc_bih, enc_bhh, c0, pH0, nullptr);
        gemm(pH0, pH1, pwEnc1, nullptr, gates, GATES4H, KTA, 0);
        lstm_pointwise_pack_kernel<<<PACK_BLOCKS, TPB, 0, stream>>>(
            gates, enc_bih + GATES4H, enc_bhh + GATES4H, c1, pH1,
            enc_outs + (size_t)t * BH);
    }

    // ---------------- Decoder ----------------
    for (int di = 0; di < SEQT; ++di) {
        dec_embed_pack_kernel<<<PACK_BLOCKS, TPB, 0, stream>>>(
            input_tensor, target_tensor, tf_flag, prevtok, di, dec_emb, pEmb);
        // attention logits: [emb, h0prev] @ Wattn^T + battn -> [B,T]
        gemm(pEmb, pH0, pwAttn, battn, attnlog, SEQT, KTA, 0);
        softmax128_kernel<<<16, 128, 0, stream>>>(attnlog, attnw);
        attn_apply_pack_kernel<<<PACK_BLOCKS, TPB, 0, stream>>>(attnw, enc_outs, pAttn);
        // combine: relu([emb, attn] @ Wcomb^T + bcomb) -> packed A fragments
        gemm(pEmb, pAttn, pwComb, bcomb, pXcomb, HID, KTA, 2);
        // LSTM layer 0
        gemm(pXcomb, pH0, pwDec0, nullptr, gates, GATES4H, KTA, 0);
        lstm_pointwise_pack_kernel<<<PACK_BLOCKS, TPB, 0, stream>>>(
            gates, dec_bih, dec_bhh, c0, pH0, nullptr);
        // LSTM layer 1
        gemm(pH0, pH1, pwDec1, nullptr, gates, GATES4H, KTA, 0);
        lstm_pointwise_pack_kernel<<<PACK_BLOCKS, TPB, 0, stream>>>(
            gates, dec_bih + GATES4H, dec_bhh + GATES4H, c1, pH1, nullptr);
        // output projection + log_softmax -> d_out[b, di, :]
        gemm(pH1, nullptr, pwOut, bout, logits, VOCAB, 0, 0);
        logsoftmax_kernel<<<16, 128, 0, stream>>>(logits, out, di);
        argmax_kernel<<<16, 128, 0, stream>>>(logits, prevtok);
    }
}
